// GATModel_77489799954881
// MI455X (gfx1250) — matrix operand order repaired and checked by reading, not compiled, not run on hardware
//
#include <hip/hip_runtime.h>
#include <hip/hip_bf16.h>
#include <math.h>

// ---------------- problem constants (from reference) ----------------
#define N_NODES 50000
#define N_EDGES 800000
#define E_TOT   850000      // edges + self loops
#define D_CH    64          // per-head channel dim
#define HD      128         // H * D
#define ED_DIM  16          // edge feature dim
#define L_NUM   3

typedef __attribute__((ext_vector_type(16))) _Float16 v16h;
typedef __attribute__((ext_vector_type(8)))  float    v8f;

// ---------------- helpers ----------------
// monotone float <-> u32 key for atomicMax-based segment max
__device__ __forceinline__ unsigned fkey(float f) {
  unsigned u = __float_as_uint(f);
  return u ^ ((unsigned)(-(int)(u >> 31)) | 0x80000000u);
}
__device__ __forceinline__ float unfkey(unsigned u) {
  unsigned m = ((u >> 31) - 1u) | 0x80000000u;
  return __uint_as_float(u ^ m);
}
// src/dst for edge e in the concatenated (orig ++ self-loop) edge list
__device__ __forceinline__ void edge_sd(const int* __restrict__ eidx, int e,
                                        int& s, int& d) {
  if (e < N_EDGES) { s = eidx[e]; d = eidx[N_EDGES + e]; }
  else             { s = e - N_EDGES; d = s; }
}
// CDNA5 async copy: 16B global -> LDS, tracked by ASYNCcnt (no VGPR data path).
// LDS offset = low 32 bits of the flat shared pointer (ISA 10.2 aperture rule).
__device__ __forceinline__ void async_g2lds_b128(void* lds_ptr, const void* gptr) {
  unsigned lo = (unsigned)(uintptr_t)lds_ptr;
  asm volatile("global_load_async_to_lds_b128 %0, %1, off"
               :: "v"(lo), "v"(gptr) : "memory");
}
__device__ __forceinline__ void wait_asynccnt0() {
  asm volatile("s_wait_asynccnt 0x0" ::: "memory");
}

// ---------------- one-time prep kernels ----------------
// transpose weights [L,64,128] f32 -> [L,128,64] f16 (column-contiguous for WMMA B)
__global__ void k_wt_transpose(const float* __restrict__ W, _Float16* __restrict__ Wt) {
  int idx = blockIdx.x * blockDim.x + threadIdx.x;
  if (idx >= L_NUM * HD * D_CH) return;
  int l   = idx / (HD * D_CH);
  int rem = idx - l * HD * D_CH;
  int n   = rem / D_CH;
  int k   = rem - n * D_CH;
  Wt[idx] = (_Float16)W[l * D_CH * HD + k * HD + n];
}

// copy edge_feat into eattr[0:E], zero self-loop region and deg
__global__ void k_prep_eattr(const float* __restrict__ efeat,
                             float* __restrict__ eattr, float* __restrict__ deg) {
  int idx = blockIdx.x * blockDim.x + threadIdx.x;
  if (idx < E_TOT * ED_DIM)
    eattr[idx] = (idx < N_EDGES * ED_DIM) ? efeat[idx] : 0.0f;
  if (idx < N_NODES) deg[idx] = 0.0f;
}

// segment-sum edge attrs into self-loop slots + in-degree counts
__global__ void k_loop_sum(const int* __restrict__ eidx, const float* __restrict__ efeat,
                           float* __restrict__ eattr, float* __restrict__ deg) {
  int idx = blockIdx.x * blockDim.x + threadIdx.x;
  if (idx >= N_EDGES * ED_DIM) return;
  int e = idx >> 4, k = idx & 15;
  int d = eidx[N_EDGES + e];
  atomicAdd(&eattr[(N_EDGES + d) * ED_DIM + k], efeat[idx]);
  if (k == 0) atomicAdd(&deg[d], 1.0f);
}

__global__ void k_loop_div(float* __restrict__ eattr, const float* __restrict__ deg) {
  int idx = blockIdx.x * blockDim.x + threadIdx.x;
  if (idx >= N_NODES * ED_DIM) return;
  int n = idx >> 4;
  eattr[N_EDGES * ED_DIM + idx] /= fmaxf(deg[n], 1.0f);
}

// ---------------- per-layer kernels ----------------
__global__ void k_tohalf(const float* __restrict__ x, _Float16* __restrict__ xh) {
  int idx = blockIdx.x * blockDim.x + threadIdx.x;
  if (idx < N_NODES * D_CH) xh[idx] = (_Float16)x[idx];
}

// Fused: YL = X@Wl + bl, YR = X@Wr + br, via v_wmma_f32_16x16x32_f16.
// One wave per 16x16 output tile; col-tiles 0..7 -> YL, 8..15 -> YR.
// Wt buffers are W pre-transposed to [128,64] f16 so B columns are contiguous.
__global__ void k_gemm_wmma(const _Float16* __restrict__ X,
                            const _Float16* __restrict__ WtL,
                            const _Float16* __restrict__ WtR,
                            const float* __restrict__ biasL,
                            const float* __restrict__ biasR,
                            float* __restrict__ YL,
                            float* __restrict__ YR) {
  int gtid = blockIdx.x * blockDim.x + threadIdx.x;
  int wave = gtid >> 5;
  int lane = threadIdx.x & 31;
  int mt = wave >> 4;              // row tile (N_NODES/16 = 3125)
  int nt = wave & 15;              // col tile across [Wl | Wr]
  if (mt >= N_NODES / 16) return;
  const _Float16* Wt   = (nt < 8) ? WtL : WtR;
  const float*    bias = (nt < 8) ? biasL : biasR;
  float*          Y    = (nt < 8) ? YL : YR;
  int ntc  = nt & 7;
  int r    = lane & 15;
  int half = lane >> 4;
  const _Float16* arow = X  + (mt * 16 + r) * D_CH;    // A: row r, 64 contiguous K
  const _Float16* bcol = Wt + (ntc * 16 + r) * D_CH;   // B: column r, 64 contiguous K
  v8f c = {};
#pragma unroll
  for (int t = 0; t < 2; ++t) {
    int kb = t * 32;
    v16h a, b;
    float4* ap = reinterpret_cast<float4*>(&a);
    float4* bp = reinterpret_cast<float4*>(&b);
    // A 16x32 layout: lane element j -> K = j + 8*((j>>3)+half)  (two 8-elem chunks)
    ap[0] = *reinterpret_cast<const float4*>(arow + kb + 8 * half);
    ap[1] = *reinterpret_cast<const float4*>(arow + kb + 16 + 8 * half);
    // B 32x16 layout: lane element j -> K = j + 8*((j>>3)+half)  (two 8-elem chunks)
    bp[0] = *reinterpret_cast<const float4*>(bcol + kb + 8 * half);
    bp[1] = *reinterpret_cast<const float4*>(bcol + kb + 16 + 8 * half);
    c = __builtin_amdgcn_wmma_f32_16x16x32_f16(false, a, false, b, (short)0, c,
                                               false, false);
  }
  float bv = bias[ntc * 16 + r];
#pragma unroll
  for (int v = 0; v < 8; ++v) {
    int row = mt * 16 + v + 8 * half;           // C/D: M = v + 8*half, N = r
    Y[row * HD + ntc * 16 + r] = c[v] + bv;
  }
}

__global__ void k_init(float* __restrict__ accum, unsigned* __restrict__ mmax,
                       float* __restrict__ denom) {
  int idx = blockIdx.x * blockDim.x + threadIdx.x;
  if (idx < N_NODES * D_CH) accum[idx] = 0.0f;
  if (idx < N_NODES * 2) { mmax[idx] = 0x007FFFFFu; /* fkey(-inf) */ denom[idx] = 0.0f; }
}

// logits[e,h] = sum_d lrelu(xl[s,h,d] + xr[d_,h,d] + (eattr[e]@We)[h,d]) * att[h,d]
// one wave per edge; ep computed on the fly from LDS-resident We (16x128).
// We/att staged into LDS with async global->LDS b128 copies (ASYNCcnt).
__global__ void k_edge_logits(const int* __restrict__ eidx,
                              const float* __restrict__ eattr,
                              const float* __restrict__ xl,
                              const float* __restrict__ xr,
                              const float* __restrict__ We,   // [16,128] layer slice
                              const float* __restrict__ att,  // [2,64]
                              float* __restrict__ logits,
                              unsigned* __restrict__ mmax) {
  __shared__ __align__(16) float sWe[ED_DIM * HD];   // 8 KB
  __shared__ __align__(16) float satt[HD];           // 512 B
  {
    const float4* gWe = reinterpret_cast<const float4*>(We);
    for (int i = threadIdx.x; i < (ED_DIM * HD) / 4; i += blockDim.x)
      async_g2lds_b128(&sWe[i * 4], gWe + i);
    const float4* gat = reinterpret_cast<const float4*>(att);
    for (int i = threadIdx.x; i < HD / 4; i += blockDim.x)
      async_g2lds_b128(&satt[i * 4], gat + i);
    wait_asynccnt0();
  }
  __syncthreads();
  int wid  = (blockIdx.x * blockDim.x + threadIdx.x) >> 5;
  int lane = threadIdx.x & 31;
  if (wid >= E_TOT) return;
  int s, d; edge_sd(eidx, wid, s, d);
  int sb = s * HD, db = d * HD;
  __builtin_prefetch(&xl[sb + lane], 0, 3);      // warm gathered rows early
  __builtin_prefetch(&xr[db + lane], 0, 3);
  float ea = (lane < ED_DIM) ? eattr[wid * ED_DIM + lane] : 0.0f;
  float acc0 = 0.0f, acc1 = 0.0f;
#pragma unroll
  for (int t = 0; t < 2; ++t) {
    int dc = lane + t * 32;                     // d index within head
    float ep0 = 0.0f, ep1 = 0.0f;
#pragma unroll
    for (int k = 0; k < ED_DIM; ++k) {
      float eak = __shfl(ea, k, 32);
      ep0 = fmaf(eak, sWe[k * HD + dc], ep0);
      ep1 = fmaf(eak, sWe[k * HD + 64 + dc], ep1);
    }
    float v0 = xl[sb + dc]      + xr[db + dc]      + ep0;
    float v1 = xl[sb + 64 + dc] + xr[db + 64 + dc] + ep1;
    v0 = v0 > 0.0f ? v0 : 0.2f * v0;            // leaky_relu(0.2)
    v1 = v1 > 0.0f ? v1 : 0.2f * v1;
    acc0 = fmaf(v0, satt[dc], acc0);
    acc1 = fmaf(v1, satt[64 + dc], acc1);
  }
#pragma unroll
  for (int o = 16; o; o >>= 1) {
    acc0 += __shfl_xor(acc0, o, 32);
    acc1 += __shfl_xor(acc1, o, 32);
  }
  if (lane == 0) {
    logits[wid * 2]     = acc0;
    logits[wid * 2 + 1] = acc1;
    atomicMax(&mmax[d * 2],     fkey(acc0));
    atomicMax(&mmax[d * 2 + 1], fkey(acc1));
  }
}

// a = exp(logit - m[dst]); in-place on logits; denom segment-sum
__global__ void k_edge_exp(const int* __restrict__ eidx,
                           float* __restrict__ logits,
                           const unsigned* __restrict__ mmax,
                           float* __restrict__ denom) {
  int idx = blockIdx.x * blockDim.x + threadIdx.x;
  if (idx >= E_TOT * 2) return;
  int e = idx >> 1, h = idx & 1;
  int s, d; edge_sd(eidx, e, s, d); (void)s;
  float m = unfkey(mmax[d * 2 + h]);
  if (!isfinite(m)) m = 0.0f;
  float a = __expf(logits[idx] - m);
  logits[idx] = a;
  atomicAdd(&denom[d * 2 + h], a);
}

// alpha = a/denom[dst]; scatter head-mean of xl[src]*alpha into accum[dst,64]
__global__ void k_edge_scatter(const int* __restrict__ eidx,
                               const float* __restrict__ xl,
                               const float* __restrict__ aexp,
                               const float* __restrict__ denom,
                               float* __restrict__ alpha_out,
                               float* __restrict__ accum) {
  int wid  = (blockIdx.x * blockDim.x + threadIdx.x) >> 5;
  int lane = threadIdx.x & 31;
  if (wid >= E_TOT) return;
  int s, d; edge_sd(eidx, wid, s, d);
  int sb = s * HD, db = d * D_CH;
  __builtin_prefetch(&xl[sb + lane], 0, 3);
  float a0 = aexp[wid * 2]     / denom[d * 2];
  float a1 = aexp[wid * 2 + 1] / denom[d * 2 + 1];
  if (lane == 0) { alpha_out[wid * 2] = a0; alpha_out[wid * 2 + 1] = a1; }
#pragma unroll
  for (int t = 0; t < 2; ++t) {
    int dc = lane + t * 32;
    float contrib = 0.5f * (xl[sb + dc] * a0 + xl[sb + 64 + dc] * a1);  // head mean
    atomicAdd(&accum[db + dc], contrib);
  }
}

// out = gelu(layernorm(accum + bias)) (+ residual); one wave per node
__global__ void k_finalize(const float* __restrict__ accum,
                           const float* __restrict__ bias,
                           const float* __restrict__ g,
                           const float* __restrict__ b,
                           const float* __restrict__ xres,
                           int do_res,
                           float* __restrict__ y) {
  int wid  = (blockIdx.x * blockDim.x + threadIdx.x) >> 5;
  int lane = threadIdx.x & 31;
  if (wid >= N_NODES) return;
  int base = wid * D_CH;
  float v0 = accum[base + lane]      + bias[lane];
  float v1 = accum[base + lane + 32] + bias[lane + 32];
  float s = v0 + v1;
#pragma unroll
  for (int o = 16; o; o >>= 1) s += __shfl_xor(s, o, 32);
  float mu = s * (1.0f / 64.0f);
  float d0 = v0 - mu, d1 = v1 - mu;
  float q = d0 * d0 + d1 * d1;
#pragma unroll
  for (int o = 16; o; o >>= 1) q += __shfl_xor(q, o, 32);
  float inv = rsqrtf(q * (1.0f / 64.0f) + 1e-5f);
  float y0 = d0 * inv * g[lane]      + b[lane];
  float y1 = d1 * inv * g[lane + 32] + b[lane + 32];
  y0 = 0.5f * y0 * (1.0f + erff(y0 * 0.70710678118654752f));   // exact gelu
  y1 = 0.5f * y1 * (1.0f + erff(y1 * 0.70710678118654752f));
  if (do_res) { y0 += xres[base + lane]; y1 += xres[base + lane + 32]; }
  y[base + lane]      = y0;
  y[base + lane + 32] = y1;
}

// ---------------- host-side launch ----------------
static inline size_t al256(size_t v) { return (v + 255) & ~(size_t)255; }

extern "C" void kernel_launch(void* const* d_in, const int* in_sizes, int n_in,
                              void* d_out, int out_size, void* d_ws, size_t ws_size,
                              hipStream_t stream) {
  const float* x0    = (const float*)d_in[0];   // [N,64]
  const int*   eidx  = (const int*)  d_in[1];   // [2,E]
  const float* efeat = (const float*)d_in[2];   // [E,16]
  const float* Wl    = (const float*)d_in[3];   // [L,64,128]
  const float* Wr    = (const float*)d_in[4];
  const float* bl    = (const float*)d_in[5];   // [L,128]
  const float* br    = (const float*)d_in[6];
  const float* We    = (const float*)d_in[7];   // [L,16,128]
  const float* att   = (const float*)d_in[8];   // [L,2,64]
  const float* bias  = (const float*)d_in[9];   // [L,64]
  const float* lng   = (const float*)d_in[10];  // [L,64]
  const float* lnb   = (const float*)d_in[11];  // [L,64]

  float* outp   = (float*)d_out;                 // [N,64]
  float* alphap = outp + (size_t)N_NODES * D_CH; // [E_TOT,2]

  // workspace carve-up
  char* p = (char*)d_ws;
  size_t off = 0;
  auto carve = [&](size_t bytes) { char* q = p + off; off = al256(off + bytes); return q; };
  float*    x_a   = (float*)   carve((size_t)N_NODES * D_CH * 4);
  float*    x_b   = (float*)   carve((size_t)N_NODES * D_CH * 4);
  _Float16* xh    = (_Float16*)carve((size_t)N_NODES * D_CH * 2);
  float*    xl    = (float*)   carve((size_t)N_NODES * HD * 4);
  float*    xr    = (float*)   carve((size_t)N_NODES * HD * 4);
  float*    eattr = (float*)   carve((size_t)E_TOT * ED_DIM * 4);
  float*    deg   = (float*)   carve((size_t)N_NODES * 4);
  unsigned* mmax  = (unsigned*)carve((size_t)N_NODES * 2 * 4);
  float*    denom = (float*)   carve((size_t)N_NODES * 2 * 4);
  float*    accum = (float*)   carve((size_t)N_NODES * D_CH * 4);
  float*    lgbuf = (float*)   carve((size_t)E_TOT * 2 * 4);
  _Float16* wlt   = (_Float16*)carve((size_t)L_NUM * HD * D_CH * 2);
  _Float16* wrt   = (_Float16*)carve((size_t)L_NUM * HD * D_CH * 2);
  (void)ws_size; (void)n_in; (void)in_sizes; (void)out_size;

  const int B = 256;
  // one-time prep
  k_wt_transpose<<<(L_NUM * HD * D_CH + B - 1) / B, B, 0, stream>>>(Wl, wlt);
  k_wt_transpose<<<(L_NUM * HD * D_CH + B - 1) / B, B, 0, stream>>>(Wr, wrt);
  k_prep_eattr<<<(E_TOT * ED_DIM + B - 1) / B, B, 0, stream>>>(efeat, eattr, deg);
  k_loop_sum<<<(N_EDGES * ED_DIM + B - 1) / B, B, 0, stream>>>(eidx, efeat, eattr, deg);
  k_loop_div<<<(N_NODES * ED_DIM + B - 1) / B, B, 0, stream>>>(eattr, deg);

  for (int l = 0; l < L_NUM; ++l) {
    const float* xsrc = (l == 0) ? x0 : ((l == 1) ? x_a : x_b);
    float*       ydst = (l == 0) ? x_a : ((l == 1) ? x_b : outp);

    k_tohalf<<<(N_NODES * D_CH + B - 1) / B, B, 0, stream>>>(xsrc, xh);
    // 3125 row-tiles * 16 col-tiles = 50000 waves, 8 waves per block
    k_gemm_wmma<<<(N_NODES / 16) * 2, B, 0, stream>>>(
        xh, wlt + (size_t)l * HD * D_CH, wrt + (size_t)l * HD * D_CH,
        bl + l * HD, br + l * HD, xl, xr);
    k_init<<<(N_NODES * D_CH + B - 1) / B, B, 0, stream>>>(accum, mmax, denom);
    k_edge_logits<<<(E_TOT + 7) / 8, B, 0, stream>>>(eidx, eattr, xl, xr,
                                                     We + (size_t)l * ED_DIM * HD,
                                                     att + (size_t)l * HD, lgbuf, mmax);
    k_edge_exp<<<(E_TOT * 2 + B - 1) / B, B, 0, stream>>>(eidx, lgbuf, mmax, denom);
    k_edge_scatter<<<(E_TOT + 7) / 8, B, 0, stream>>>(eidx, xl, lgbuf, denom,
                                                      alphap, accum);
    k_finalize<<<(N_NODES + 7) / 8, B, 0, stream>>>(accum, bias + l * D_CH,
                                                    lng + l * D_CH, lnb + l * D_CH,
                                                    xsrc, (l < 2) ? 1 : 0, ydst);
  }
}
